// MultiHeadAttention_72413148611107
// MI455X (gfx1250) — matrix operand order repaired and checked by reading, not compiled, not run on hardware
//
#include <hip/hip_runtime.h>

// ---------------------------------------------------------------------------
// MHA forward for MI455X (gfx1250, wave32, WMMA).
// B=2, S=2048, D=1024, H=16, HD=64.  All GEMMs on v_wmma_f32_16x16x32_bf16.
// Round 3: fix async-to-LDS builtin argument types (int4-vector pointers in
//          AS1/AS3, per hipcc diagnostic).
// ---------------------------------------------------------------------------

#define BATCH 2
#define SEQ   2048
#define DIM   1024
#define NH    16
#define HDIM  64
#define MROWS (BATCH * SEQ)       // 4096

typedef __attribute__((ext_vector_type(16))) __bf16 v16bf;
typedef __attribute__((ext_vector_type(8)))  float  v8f;
typedef __attribute__((ext_vector_type(4)))  unsigned int u32x4;
typedef int gv4i __attribute__((vector_size(16)));   // matches builtin param type

union Frag16 {            // 32 bytes: one WMMA bf16 A or B operand (8 VGPRs)
    u32x4 q[2];
    v16bf v;
    unsigned short s[16];
};

static __device__ inline unsigned short f2bf(float f) {
    unsigned int u = __float_as_uint(f);
    u += 0x7FFFu + ((u >> 16) & 1u);          // round-to-nearest-even
    return (unsigned short)(u >> 16);
}

static __device__ inline v8f wmma_bf16(const Frag16& a, const Frag16& b, v8f c) {
    return __builtin_amdgcn_wmma_f32_16x16x32_bf16(
        /*neg_a=*/false, a.v, /*neg_b=*/false, b.v,
        /*c_mod=*/(short)0, c, /*reuse_a=*/false, /*reuse_b=*/false);
}

// --------------------------- async global->LDS ------------------------------
#if defined(__has_builtin)
#if __has_builtin(__builtin_amdgcn_global_load_async_to_lds_b128) && \
    __has_builtin(__builtin_amdgcn_s_wait_asynccnt)
#define HAVE_ASYNC_LDS 1
#endif
#endif

static __device__ inline void lds_copy16_issue(const unsigned short* g, unsigned short* l) {
#ifdef HAVE_ASYNC_LDS
    __builtin_amdgcn_global_load_async_to_lds_b128(
        (__attribute__((address_space(1))) gv4i*)(unsigned long long)(const void*)g,
        (__attribute__((address_space(3))) gv4i*)(unsigned int)(unsigned long long)(void*)l,
        0, 0);
#else
    *(u32x4*)l = *(const u32x4*)g;
#endif
}
static __device__ inline void lds_copy_wait() {
#ifdef HAVE_ASYNC_LDS
    __builtin_amdgcn_s_wait_asynccnt(0);
#endif
}

// ---------------------------------------------------------------------------
// f32 -> bf16 conversion (4 elements / thread)
// ---------------------------------------------------------------------------
__global__ __launch_bounds__(256)
void convert_f32_bf16(const float* __restrict__ src, unsigned short* __restrict__ dst, int n) {
    int i4 = (blockIdx.x * blockDim.x + threadIdx.x) * 4;
    if (i4 >= n) return;
    float4 f = *(const float4*)(src + i4);
    uint2 o;
    o.x = (unsigned)f2bf(f.x) | ((unsigned)f2bf(f.y) << 16);
    o.y = (unsigned)f2bf(f.z) | ((unsigned)f2bf(f.w) << 16);
    *(uint2*)(dst + i4) = o;
}

// ---------------------------------------------------------------------------
// GEMM: out[m,n] = sum_k A[m,k] * W[n,k]   (A: MROWS x DIM, W: DIM x DIM, bf16)
// Block tile 128x256, BK=32, 256 threads = 8 waves (2x4), wave tile 64x64.
// Double-buffered async staging. out_f32==0 -> bf16 out, ==1 -> f32 out + bias.
// ---------------------------------------------------------------------------
#define LSTR 40   // padded LDS row stride in ushorts (80 B, 16-B aligned)

__global__ __launch_bounds__(256)
void gemm_bf16_nt(const unsigned short* __restrict__ A,
                  const unsigned short* __restrict__ W,
                  void* __restrict__ out,
                  const float* __restrict__ bias,
                  int out_f32) {
    __shared__ unsigned short As[2][128 * LSTR];
    __shared__ unsigned short Bs[2][256 * LSTR];

    const int bm = blockIdx.y * 128;
    const int bn = blockIdx.x * 256;
    const int tid  = threadIdx.x;
    const int lane = tid & 31;
    const int wave = tid >> 5;
    const int wm = wave >> 2;        // 0..1  -> 64 rows
    const int wn = wave & 3;         // 0..3  -> 64 cols
    const int ln = lane & 15;
    const int hh = lane >> 4;

    v8f acc[4][4];
    #pragma unroll
    for (int i = 0; i < 4; ++i)
        #pragma unroll
        for (int j = 0; j < 4; ++j)
            acc[i][j] = (v8f){0.f,0.f,0.f,0.f,0.f,0.f,0.f,0.f};

    // staging: 16-B chunks; A has 512 chunk-slots, B has 1024
    auto issue_tile = [&](int kb, int buf) {
        #pragma unroll
        for (int u = 0; u < 2; ++u) {
            const int c = tid + u * 256;
            const int row = c >> 2, off = (c & 3) * 8;
            lds_copy16_issue(A + (size_t)(bm + row) * DIM + kb + off,
                             &As[buf][row * LSTR + off]);
        }
        #pragma unroll
        for (int u = 0; u < 4; ++u) {
            const int c = tid + u * 256;
            const int row = c >> 2, off = (c & 3) * 8;
            lds_copy16_issue(W + (size_t)(bn + row) * DIM + kb + off,
                             &Bs[buf][row * LSTR + off]);
        }
    };

    issue_tile(0, 0);

    const int NKB = DIM / 32;
    for (int kb = 0; kb < NKB; ++kb) {
        const int cur = kb & 1;
        lds_copy_wait();           // tile kb resident (this wave's issues)
        __syncthreads();           // all waves: tile kb visible, prev compute done
        if (kb + 1 < NKB) issue_tile((kb + 1) * 32, 1 - cur);

        Frag16 af[4], bf4[4];
        #pragma unroll
        for (int i = 0; i < 4; ++i) {
            const int r = wm * 64 + i * 16 + ln;
            af[i].q[0] = *(const u32x4*)&As[cur][r * LSTR + hh * 8];
            af[i].q[1] = *(const u32x4*)&As[cur][r * LSTR + 16 + hh * 8];
        }
        #pragma unroll
        for (int j = 0; j < 4; ++j) {
            const int r = wn * 64 + j * 16 + ln;
            bf4[j].q[0] = *(const u32x4*)&Bs[cur][r * LSTR + hh * 8];
            bf4[j].q[1] = *(const u32x4*)&Bs[cur][r * LSTR + 16 + hh * 8];
        }
        #pragma unroll
        for (int i = 0; i < 4; ++i)
            #pragma unroll
            for (int j = 0; j < 4; ++j)
                acc[i][j] = wmma_bf16(af[i], bf4[j], acc[i][j]);
    }

    // store: C layout -> lane col n=ln, rows hh*8 + r
    #pragma unroll
    for (int i = 0; i < 4; ++i) {
        #pragma unroll
        for (int j = 0; j < 4; ++j) {
            const int col  = bn + wn * 64 + j * 16 + ln;
            const int row0 = bm + wm * 64 + i * 16 + hh * 8;
            const float bv = bias ? bias[col] : 0.f;
            #pragma unroll
            for (int r = 0; r < 8; ++r) {
                const float v = acc[i][j][r] + bv;
                if (out_f32)
                    ((float*)out)[(size_t)(row0 + r) * DIM + col] = v;
                else
                    ((unsigned short*)out)[(size_t)(row0 + r) * DIM + col] = f2bf(v);
            }
        }
    }
}

// ---------------------------------------------------------------------------
// Flash attention: grid (S/64, B*H), block 128 (4 waves, 16 query rows each).
// 64-key blocks. Q/K fragments direct from global (K-contiguous per lane);
// V staged transposed in LDS; P round-trips C-layout -> A-layout via LDS.
// ---------------------------------------------------------------------------
#define KBLK 64
#define VSTR 72   // 64 keys + 8 pad (144 B rows, 16-B aligned)

__global__ __launch_bounds__(128)
void flash_attn(const unsigned short* __restrict__ Q,
                const unsigned short* __restrict__ K,
                const unsigned short* __restrict__ V,
                unsigned short* __restrict__ ctx) {
    __shared__ unsigned short Vt[HDIM * VSTR];      // [hd][key], padded
    __shared__ unsigned short Pb[4][16 * VSTR];     // per-wave P tile [16][64]

    const int b  = blockIdx.y >> 4;
    const int h  = blockIdx.y & 15;
    const int qb = blockIdx.x;
    const int tid  = threadIdx.x;
    const int lane = tid & 31;
    const int wave = tid >> 5;
    const int ln = lane & 15;
    const int hh = lane >> 4;

    const size_t bh = (size_t)b * SEQ * DIM + h * HDIM;   // (b, s=0, h, hd=0)
    const int qs0 = qb * 64 + wave * 16;

    // Q fragments (2 K-steps of 32 over HD=64), loaded once
    Frag16 qf[2];
    #pragma unroll
    for (int kk = 0; kk < 2; ++kk) {
        const unsigned short* qp = Q + bh + (size_t)(qs0 + ln) * DIM + kk * 32;
        qf[kk].q[0] = *(const u32x4*)(qp + hh * 8);
        qf[kk].q[1] = *(const u32x4*)(qp + 16 + hh * 8);
    }

    v8f co[4];
    #pragma unroll
    for (int t = 0; t < 4; ++t) co[t] = (v8f){0.f,0.f,0.f,0.f,0.f,0.f,0.f,0.f};
    float mrow[8], lrow[8];
    #pragma unroll
    for (int r = 0; r < 8; ++r) { mrow[r] = -__builtin_inff(); lrow[r] = 0.f; }

    const int nkb = qb + 1;    // causal bound, 64-key blocks

    for (int kb = 0; kb < nkb; ++kb) {
        const int ks0 = kb * KBLK;
        __syncthreads();   // protect Vt/Pb from previous iteration's readers

        // cooperative V stage (transposed): 128 thr * 2 chunks * 16 bf16
        #pragma unroll
        for (int c = 0; c < 2; ++c) {
            const int cid = tid * 2 + c;
            const int key = cid >> 2;              // 0..63
            const int c0  = (cid & 3) * 16;        // hd chunk
            Frag16 t;
            const u32x4* gv = (const u32x4*)(V + bh + (size_t)(ks0 + key) * DIM + c0);
            t.q[0] = gv[0]; t.q[1] = gv[1];
            #pragma unroll
            for (int e = 0; e < 16; ++e)
                Vt[(c0 + e) * VSTR + key] = t.s[e];
        }
        // prefetch next key block (K and V rows)
        if (kb + 1 < nkb) {
            __builtin_prefetch(K + bh + (size_t)(ks0 + KBLK + lane) * DIM, 0, 1);
            __builtin_prefetch(V + bh + (size_t)(ks0 + KBLK + lane) * DIM, 0, 1);
        }
        __syncthreads();

        // S = Q K^T  (four 16-col key tiles)
        v8f sa[4];
        #pragma unroll
        for (int j = 0; j < 4; ++j) sa[j] = (v8f){0.f,0.f,0.f,0.f,0.f,0.f,0.f,0.f};
        #pragma unroll
        for (int kk = 0; kk < 2; ++kk) {
            #pragma unroll
            for (int j = 0; j < 4; ++j) {
                Frag16 kf;
                const unsigned short* kp =
                    K + bh + (size_t)(ks0 + j * 16 + ln) * DIM + kk * 32 + hh * 8;
                kf.q[0] = *(const u32x4*)kp;
                kf.q[1] = *(const u32x4*)(kp + 16);
                sa[j] = wmma_bf16(qf[kk], kf, sa[j]);
            }
        }

        // scale + causal mask (C layout: col = ln, rows = hh*8 + r)
        #pragma unroll
        for (int j = 0; j < 4; ++j) {
            const int col = ks0 + j * 16 + ln;
            #pragma unroll
            for (int r = 0; r < 8; ++r) {
                const int row = qs0 + hh * 8 + r;
                float x = sa[j][r] * 0.125f;
                sa[j][r] = (col > row) ? -__builtin_inff() : x;
            }
        }

        // online softmax stats (rows live on 16 lanes of one half-wave)
        float alpha[8];
        #pragma unroll
        for (int r = 0; r < 8; ++r) {
            float mx = fmaxf(fmaxf(sa[0][r], sa[1][r]), fmaxf(sa[2][r], sa[3][r]));
            #pragma unroll
            for (int m = 1; m < 16; m <<= 1) mx = fmaxf(mx, __shfl_xor(mx, m));
            const float mn = fmaxf(mrow[r], mx);
            alpha[r] = __expf(mrow[r] - mn);
            mrow[r] = mn;
        }
        #pragma unroll
        for (int j = 0; j < 4; ++j)
            #pragma unroll
            for (int r = 0; r < 8; ++r) {
                const float p = __expf(sa[j][r] - mrow[r]);
                sa[j][r] = p;
                Pb[wave][(hh * 8 + r) * VSTR + j * 16 + ln] = f2bf(p);
            }
        #pragma unroll
        for (int r = 0; r < 8; ++r) {
            float s = (sa[0][r] + sa[1][r]) + (sa[2][r] + sa[3][r]);
            #pragma unroll
            for (int m = 1; m < 16; m <<= 1) s += __shfl_xor(s, m);
            lrow[r] = alpha[r] * lrow[r] + s;
        }
        #pragma unroll
        for (int t = 0; t < 4; ++t)
            #pragma unroll
            for (int r = 0; r < 8; ++r) co[t][r] *= alpha[r];

        __syncthreads();   // make P visible before PV

        // PV: ctx += P @ V   (A from Pb, B from transposed Vt), K = 64 = 2 steps
        #pragma unroll
        for (int kk = 0; kk < 2; ++kk) {
            Frag16 pf;
            pf.q[0] = *(const u32x4*)&Pb[wave][ln * VSTR + kk * 32 + hh * 8];
            pf.q[1] = *(const u32x4*)&Pb[wave][ln * VSTR + kk * 32 + 16 + hh * 8];
            #pragma unroll
            for (int t = 0; t < 4; ++t) {
                Frag16 vf;
                vf.q[0] = *(const u32x4*)&Vt[(t * 16 + ln) * VSTR + kk * 32 + hh * 8];
                vf.q[1] = *(const u32x4*)&Vt[(t * 16 + ln) * VSTR + kk * 32 + 16 + hh * 8];
                co[t] = wmma_bf16(pf, vf, co[t]);
            }
        }
    }

    // epilogue: ctx /= l, store bf16 [b, s, h, hd]
    #pragma unroll
    for (int r = 0; r < 8; ++r) {
        const float inv = 1.0f / lrow[r];
        const size_t row = bh + (size_t)(qs0 + hh * 8 + r) * DIM;
        #pragma unroll
        for (int t = 0; t < 4; ++t)
            ctx[row + t * 16 + ln] = f2bf(co[t][r] * inv);
    }
}

// ---------------------------------------------------------------------------
// Host-side launcher
// ---------------------------------------------------------------------------
extern "C" void kernel_launch(void* const* d_in, const int* in_sizes, int n_in,
                              void* d_out, int out_size, void* d_ws, size_t ws_size,
                              hipStream_t stream) {
    const float* x  = (const float*)d_in[0];
    const float* Wq = (const float*)d_in[1];
    const float* Wk = (const float*)d_in[2];
    const float* Wv = (const float*)d_in[3];
    const float* Wo = (const float*)d_in[4];
    const float* bo = (const float*)d_in[5];

    char* w = (char*)d_ws;
    const size_t XB = (size_t)MROWS * DIM * 2;   // 8 MB bf16
    const size_t WB = (size_t)DIM * DIM * 2;     // 2 MB bf16
    unsigned short* xb   = (unsigned short*)(w);
    unsigned short* wqb  = (unsigned short*)(w + XB);
    unsigned short* wkb  = (unsigned short*)(w + XB + WB);
    unsigned short* wvb  = (unsigned short*)(w + XB + 2 * WB);
    unsigned short* wob  = (unsigned short*)(w + XB + 3 * WB);
    unsigned short* qb   = (unsigned short*)(w + XB + 4 * WB);
    unsigned short* kbuf = (unsigned short*)(w + 2 * XB + 4 * WB);
    unsigned short* vbuf = (unsigned short*)(w + 3 * XB + 4 * WB);
    unsigned short* cbuf = (unsigned short*)(w + 4 * XB + 4 * WB);

    const int nx = MROWS * DIM;      // 4194304
    const int nw = DIM * DIM;        // 1048576
    convert_f32_bf16<<<nx / 4 / 256, 256, 0, stream>>>(x,  xb,  nx);
    convert_f32_bf16<<<nw / 4 / 256, 256, 0, stream>>>(Wq, wqb, nw);
    convert_f32_bf16<<<nw / 4 / 256, 256, 0, stream>>>(Wk, wkb, nw);
    convert_f32_bf16<<<nw / 4 / 256, 256, 0, stream>>>(Wv, wvb, nw);
    convert_f32_bf16<<<nw / 4 / 256, 256, 0, stream>>>(Wo, wob, nw);

    dim3 ggrid(DIM / 256, MROWS / 128);          // (4, 32)
    gemm_bf16_nt<<<ggrid, 256, 0, stream>>>(xb, wqb, qb,   nullptr, 0);
    gemm_bf16_nt<<<ggrid, 256, 0, stream>>>(xb, wkb, kbuf, nullptr, 0);
    gemm_bf16_nt<<<ggrid, 256, 0, stream>>>(xb, wvb, vbuf, nullptr, 0);

    dim3 fgrid(SEQ / 64, BATCH * NH);            // (32, 32)
    flash_attn<<<fgrid, 128, 0, stream>>>(qb, kbuf, vbuf, cbuf);

    gemm_bf16_nt<<<ggrid, 256, 0, stream>>>(cbuf, wob, d_out, bo, 1);
}
